// TokenEmbeddingGCN_72043781423517
// MI455X (gfx1250) — compile-verified, matrix operand order checked by reading, not run on hardware
//
#include <hip/hip_runtime.h>
#include <hip/hip_bf16.h>

typedef __bf16 bf16_t;
typedef bf16_t v8bf  __attribute__((ext_vector_type(8)));
typedef bf16_t v16bf __attribute__((ext_vector_type(16)));
typedef float  v8f   __attribute__((ext_vector_type(8)));

#define B_   8
#define N_   1024
#define P_   16
#define D_   512
#define E_   8192           // P_*D_
#define M_   8192           // B_*N_

// ---------------------------------------------------------------------------
// Kernel 0: transpose + convert W2 (D_ x E_, f32 row-major) -> W2T (E_ x D_, bf16)
// LDS 32x33 tile transpose: coalesced reads along e, coalesced writes along k.
// ---------------------------------------------------------------------------
__global__ __launch_bounds__(256) void w2_transpose(const float* __restrict__ W2,
                                                    bf16_t* __restrict__ W2T) {
    __shared__ float tile[32][33];
    const int ebase = blockIdx.x * 32;     // along E_
    const int kbase = blockIdx.y * 32;     // along D_
    const int tx = threadIdx.x & 31;
    const int ty = threadIdx.x >> 5;       // 0..7
#pragma unroll
    for (int i = 0; i < 4; ++i)
        tile[ty + 8 * i][tx] = W2[(size_t)(kbase + ty + 8 * i) * E_ + ebase + tx];
    __syncthreads();
#pragma unroll
    for (int i = 0; i < 4; ++i)
        W2T[(size_t)(ebase + ty + 8 * i) * D_ + kbase + tx] = (bf16_t)tile[tx][ty + 8 * i];
}

// ---------------------------------------------------------------------------
// Kernel 1: h = relu(xp @ W1 + b1), stored bf16.  One block per row (8192 rows).
// K=16 only -> plain VALU is fine (0.13 GFLOP total).
// ---------------------------------------------------------------------------
__global__ __launch_bounds__(256) void gen_h(const float* __restrict__ x,
                                             const float* __restrict__ W1,
                                             const float* __restrict__ b1,
                                             bf16_t* __restrict__ h) {
    __shared__ float xs[P_];
    const int row = blockIdx.x;            // 0..M_-1
    if (threadIdx.x < P_) xs[threadIdx.x] = x[(size_t)row * P_ + threadIdx.x];
    __syncthreads();
#pragma unroll
    for (int j = 0; j < 2; ++j) {
        const int d = threadIdx.x + j * 256;
        float acc = b1[d];
#pragma unroll
        for (int p = 0; p < P_; ++p) acc += xs[p] * W1[p * D_ + d];
        acc = fmaxf(acc, 0.0f);
        h[(size_t)row * D_ + d] = (bf16_t)acc;
    }
}

// ---------------------------------------------------------------------------
// Kernel 2: filt = relu(h @ W2 + b2)   (M_ x E_), f32 out, bf16 WMMA compute.
// Wave computes a 16x64 output tile (4 accumulators). 8 waves / block.
// A/B fragment layout per CDNA5 ISA 16-bit tables:
//   lane = half*16 + m; VGPR v holds K = (v/4)*16 + half*8 + (v%4)*2 + {0,1}
//   -> per lane: two contiguous 8-element (16B) chunks at k+half*8 and k+16+half*8.
// ---------------------------------------------------------------------------
__global__ __launch_bounds__(256) void gemm2(const bf16_t* __restrict__ h,
                                             const bf16_t* __restrict__ w2t,
                                             const float* __restrict__ b2,
                                             float* __restrict__ filt) {
    const int lane = threadIdx.x & 31;
    const int wave = threadIdx.x >> 5;
    const int unit = blockIdx.x * 8 + wave;      // 0..65535
    const int tileM  = unit >> 7;                // 0..511   (16 rows each)
    const int tileEg = unit & 127;               // 0..127   (64 cols each)
    const int m16  = lane & 15;
    const int half = lane >> 4;

    const bf16_t* __restrict__ arow  = h   + (size_t)(tileM * 16 + m16) * D_ + half * 8;
    const bf16_t* __restrict__ brow0 = w2t + (size_t)(tileEg * 64 + m16) * D_ + half * 8;

    v8f acc[4] = {v8f{}, v8f{}, v8f{}, v8f{}};

    for (int k = 0; k < D_; k += 32) {
        v8bf a0 = *(const v8bf*)(arow + k);
        v8bf a1 = *(const v8bf*)(arow + k + 16);
        v16bf afrag = __builtin_shufflevector(a0, a1, 0, 1, 2, 3, 4, 5, 6, 7,
                                              8, 9, 10, 11, 12, 13, 14, 15);
#pragma unroll
        for (int t = 0; t < 4; ++t) {
            const bf16_t* bp = brow0 + (size_t)t * 16 * D_ + k;
            v8bf b0 = *(const v8bf*)(bp);
            v8bf b1 = *(const v8bf*)(bp + 16);
            v16bf bfrag = __builtin_shufflevector(b0, b1, 0, 1, 2, 3, 4, 5, 6, 7,
                                                  8, 9, 10, 11, 12, 13, 14, 15);
            acc[t] = __builtin_amdgcn_wmma_f32_16x16x32_bf16(
                false, afrag, false, bfrag, (short)0, acc[t], false, false);
        }
    }

    // Epilogue: C layout — lane half h, col n=lane&15, VGPR r -> row = r + h*8.
    const int mOut = tileM * 16 + half * 8;
#pragma unroll
    for (int t = 0; t < 4; ++t) {
        const int e = tileEg * 64 + t * 16 + m16;
        const float bias = b2[e];
#pragma unroll
        for (int r = 0; r < 8; ++r) {
            float v = acc[t][r] + bias;
            v = v > 0.0f ? v : 0.0f;
            filt[(size_t)(mOut + r) * E_ + e] = v;
        }
    }
}

// ---------------------------------------------------------------------------
// Kernel 3: online softmax stats per (b, e): max & sum(exp) over N_ rows.
// Coalesced: consecutive threads -> consecutive e.
// ---------------------------------------------------------------------------
__global__ __launch_bounds__(256) void softmax_stats(const float* __restrict__ filt,
                                                     float* __restrict__ cmax,
                                                     float* __restrict__ csum) {
    const int b = blockIdx.y;
    const int e = blockIdx.x * 256 + threadIdx.x;
    const float* __restrict__ col = filt + (size_t)b * N_ * E_ + e;
    float m = -3.402823466e38f, s = 0.0f;
    for (int n = 0; n < N_; ++n) {
        const float v = col[(size_t)n * E_];
        const float nm = fmaxf(m, v);
        s = s * __expf(m - nm) + __expf(v - nm);
        m = nm;
    }
    cmax[b * E_ + e] = m;
    csum[b * E_ + e] = s;
}

// ---------------------------------------------------------------------------
// Kernel 4: apply softmax, multiply by xp, transpose (d,p)->(p,d), avg-pool
// k=3/pad=1 over d. One block per (b, n). LDS tile padded to stride 17
// (conflict-free reads down d). Coalesced stores over d.
// ---------------------------------------------------------------------------
__global__ __launch_bounds__(256) void apply_pool(const float* __restrict__ x,
                                                  const float* __restrict__ filt,
                                                  const float* __restrict__ cmax,
                                                  const float* __restrict__ csum,
                                                  float* __restrict__ out) {
    __shared__ float xs[P_];
    __shared__ float t[D_ * 17];
    const int b = blockIdx.x >> 10;
    const int n = blockIdx.x & (N_ - 1);
    if (threadIdx.x < P_)
        xs[threadIdx.x] = x[((size_t)b * N_ + n) * P_ + threadIdx.x];
    __syncthreads();

    const float* __restrict__ frow = filt + ((size_t)b * N_ + n) * E_;
    const float* __restrict__ mb = cmax + (size_t)b * E_;
    const float* __restrict__ sb = csum + (size_t)b * E_;
    for (int e = threadIdx.x; e < E_; e += 256) {
        const float sm = __expf(frow[e] - mb[e]) / sb[e];
        t[(e >> 4) * 17 + (e & 15)] = sm * xs[e & 15];
    }
    __syncthreads();

    float* __restrict__ orow = out + ((size_t)b * (N_ * P_) + n * P_) * D_;
    for (int idx = threadIdx.x; idx < P_ * D_; idx += 256) {
        const int p = idx >> 9;          // 0..15
        const int d = idx & (D_ - 1);    // 0..511 (consecutive per lane)
        const float c = t[d * 17 + p];
        const float l = (d > 0)        ? t[(d - 1) * 17 + p] : 0.0f;
        const float r = (d < D_ - 1)   ? t[(d + 1) * 17 + p] : 0.0f;
        orow[(size_t)p * D_ + d] = (l + c + r) * (1.0f / 3.0f);
    }
}

// ---------------------------------------------------------------------------
extern "C" void kernel_launch(void* const* d_in, const int* in_sizes, int n_in,
                              void* d_out, int out_size, void* d_ws, size_t ws_size,
                              hipStream_t stream) {
    const float* x  = (const float*)d_in[0];
    const float* W1 = (const float*)d_in[1];
    const float* b1 = (const float*)d_in[2];
    const float* W2 = (const float*)d_in[3];
    const float* b2 = (const float*)d_in[4];
    float* out = (float*)d_out;

    char* ws = (char*)d_ws;
    bf16_t* h_bf   = (bf16_t*)(ws);                                  //  8 MB
    bf16_t* w2t_bf = (bf16_t*)(ws + (size_t)M_ * D_ * 2);            //  8 MB
    float*  cmax   = (float*)(ws + 2 * (size_t)M_ * D_ * 2);         //  256 KB
    float*  csum   = (float*)((char*)cmax + (size_t)B_ * E_ * 4);    //  256 KB
    float*  filt   = (float*)((char*)csum + (size_t)B_ * E_ * 4);    //  256 MB

    // 0) W2 -> bf16 transposed (E_ x D_)
    w2_transpose<<<dim3(E_ / 32, D_ / 32), 256, 0, stream>>>(W2, w2t_bf);
    // 1) h = relu(xp@W1+b1), bf16
    gen_h<<<M_, 256, 0, stream>>>(x, W1, b1, h_bf);
    // 2) filt = relu(h@W2+b2) via v_wmma_f32_16x16x32_bf16
    gemm2<<<(M_ / 16) * (E_ / 64) / 8, 256, 0, stream>>>(h_bf, w2t_bf, b2, filt);
    // 3) per-column online softmax stats
    softmax_stats<<<dim3(E_ / 256, B_), 256, 0, stream>>>(filt, cmax, csum);
    // 4) softmax apply * xp, transpose, avg-pool
    apply_pool<<<B_ * N_, 256, 0, stream>>>(x, filt, cmax, csum, out);
}
